// FP4Linear_77996606095675
// MI455X (gfx1250) — matrix operand order repaired and checked by reading, not compiled, not run on hardware
//
#include <hip/hip_runtime.h>
#include <stdint.h>

typedef __attribute__((ext_vector_type(16))) __bf16 v16bf;
typedef __attribute__((ext_vector_type(8)))  __bf16 v8bf;
typedef __attribute__((ext_vector_type(8)))  float  v8f;
typedef __attribute__((ext_vector_type(4)))  int    v4i;

#define AS1 __attribute__((address_space(1)))
#define AS3 __attribute__((address_space(3)))

#define M_TOT 512
#define N_TOT 11008
#define K_TOT 4096
#define BM 128
#define BN 128
#define KC 64
#define NKT (K_TOT / KC)
#define LDT 80   // padded LDS row stride (elements); 160B rows, 16B aligned

#if __has_builtin(__builtin_amdgcn_global_load_async_to_lds_b128)
#define HAS_ASYNC 1
#else
#define HAS_ASYNC 0
#endif

// global -> LDS, 16 bytes, no VGPR staging (ASYNCcnt-tracked on CDNA5)
__device__ __forceinline__ void async_copy_b128(const uint16_t* g, AS3 uint16_t* l) {
#if HAS_ASYNC
  __builtin_amdgcn_global_load_async_to_lds_b128(
      (AS1 v4i*)(uintptr_t)g, (AS3 v4i*)l, 0, 0);
#else
  *(AS3 v4i*)l = *reinterpret_cast<const v4i*>(g);
#endif
}

__device__ __forceinline__ void wait_async0() {
#if HAS_ASYNC
#if __has_builtin(__builtin_amdgcn_s_wait_asynccnt)
  __builtin_amdgcn_s_wait_asynccnt(0);
#else
  asm volatile("s_wait_asynccnt 0" ::: "memory");
#endif
#endif
}

// ---- e2m1 dequant ----------------------------------------------------------
// magnitudes (em = code&7): {0, .5, 1, 1.5, 2, 3, 4, 6} as bf16 bit patterns:
//   hi bytes: 00 3F 3F 3F 40 40 40 40   lo bytes: 00 00 80 C0 00 40 80 C0

// branch-free fallback (pure mask arithmetic, nothing for clang to if-convert)
__device__ __forceinline__ uint32_t nib2bf_mask(uint32_t c) {
  uint32_t em    = c & 7u;
  uint32_t small = 0u - (uint32_t)(em < 2u);      // all-ones if em < 2
  uint32_t magS  = em * 0x3F00u;                  // 0 or 0x3F00
  uint32_t magN  = (((em >> 1) + 126u) << 7) | ((em & 1u) << 6);
  uint32_t mag   = (small & magS) | (~small & magN);
  return mag | ((c & 8u) << 12);
}

// packed byte b (two e2m1 codes) -> two packed bf16 (lo code -> [15:0])
__device__ __forceinline__ uint32_t dq2(uint32_t b) {
#if __has_builtin(__builtin_amdgcn_perm)
  // selector bytes: [em_lo, em_lo, em_hi, em_hi]
  uint32_t sel = (b & 7u) * 0x0101u + ((b >> 4) & 7u) * 0x01010000u;
  uint32_t pLo = __builtin_amdgcn_perm(0xC0804000u, 0xC0800000u, sel); // mag lo-bytes
  uint32_t pHi = __builtin_amdgcn_perm(0x40404040u, 0x3F3F3F00u, sel); // mag hi-bytes
  uint32_t mag = (pHi & 0xFF00FF00u) | (pLo & 0x00FF00FFu);            // v_bfi
  return mag | ((b & 8u) << 12) | ((b & 0x80u) << 24);                 // signs
#else
  return nib2bf_mask(b & 0xFu) | (nib2bf_mask((b >> 4) & 0xFu) << 16);
#endif
}

__device__ __forceinline__ uint16_t f32_to_bf16_rne(float f) {
  uint32_t u = __builtin_bit_cast(uint32_t, f);
  u += 0x7FFFu + ((u >> 16) & 1u);
  return (uint16_t)(u >> 16);
}
__device__ __forceinline__ float bf16bits_to_f32(uint32_t b) {
  return __builtin_bit_cast(float, b << 16);
}

__global__ __launch_bounds__(256)
void fp4_linear_wmma(const uint16_t* __restrict__ x,
                     const int*      __restrict__ wp,
                     const int*      __restrict__ bp,
                     uint16_t*       __restrict__ out) {
  __shared__ uint16_t sA[2 * BM * LDT];   // x tiles (bf16 bits), double buffered
  __shared__ uint16_t sW[2 * BN * LDT];   // dequantized W tiles, double buffered

  const int tid   = threadIdx.x;
  const int lane  = tid & 31;
  const int wave  = tid >> 5;     // 0..7
  const int waveM = wave & 1;     // 2 waves along M (64 rows each)
  const int waveN = wave >> 1;    // 4 waves along N (32 cols each)

  const int blockN0 = blockIdx.x * BN;
  const int blockM0 = blockIdx.y * BM;

  // staging: thread -> (row r, half h); each half = 32 K-elements (64 B)
  const int r = tid >> 1;
  const int h = tid & 1;

  const uint16_t* xrow = x  + (size_t)(blockM0 + r) * K_TOT       + h * 32;
  const int*      wrow = wp + (size_t)(blockN0 + r) * (K_TOT / 2) + h * 16;

  // Cast ONLY the LDS base to AS(3); all offsets are AS(3) pointer arithmetic
  // (constant addrspacecast of offset GEPs is not linkable; base-only is).
  AS3 uint16_t* a3 = (AS3 uint16_t*)sA;
  AS3 uint16_t* w3 = (AS3 uint16_t*)sW;
  AS3 uint16_t* sA3[2]  = { a3, a3 + BM * LDT };
  AS3 uint16_t* sW3[2]  = { w3, w3 + BN * LDT };
  AS3 uint16_t* aDst[2] = { sA3[0] + (r * LDT + h * 32),
                            sA3[1] + (r * LDT + h * 32) };
  AS3 uint32_t* wDst[2] = { (AS3 uint32_t*)(sW3[0] + (r * LDT + h * 32)),
                            (AS3 uint32_t*)(sW3[1] + (r * LDT + h * 32)) };

  const int laneHalf = lane >> 4;
  const int lane15   = lane & 15;

  v8f acc[4][2] = {};

  // ---- prologue: stage tile 0 into buffer 0 ----
  {
    #pragma unroll
    for (int i = 0; i < 4; ++i)
      async_copy_b128(xrow + i * 8, aDst[0] + i * 8);
    int4 wq[4];
    const int4* wv = reinterpret_cast<const int4*>(wrow);
    #pragma unroll
    for (int i = 0; i < 4; ++i) wq[i] = wv[i];
    const uint32_t* q = reinterpret_cast<const uint32_t*>(wq);
    #pragma unroll
    for (int i = 0; i < 16; ++i) wDst[0][i] = dq2(q[i]);
    wait_async0();
    __syncthreads();
  }

  for (int kt = 0; kt < NKT; ++kt) {
    const int cur = kt & 1;
    const int nxt = cur ^ 1;
    const bool haveNext = (kt + 1 < NKT);

    // 1) kick off next x-tile: global -> LDS async (other buffer)
    if (haveNext) {
      const uint16_t* g = xrow + (size_t)(kt + 1) * KC;
      #pragma unroll
      for (int i = 0; i < 4; ++i)
        async_copy_b128(g + i * 8, aDst[nxt] + i * 8);
    }
    // 2) kick off next W-tile packed loads (16 VGPRs live across compute)
    int4 wq[4];
    if (haveNext) {
      const int4* wv = reinterpret_cast<const int4*>(wrow + (kt + 1) * (KC / 2));
      #pragma unroll
      for (int i = 0; i < 4; ++i) wq[i] = wv[i];
    }

    // 3) compute current tile: two 16x16x32 K-steps
    AS3 const uint16_t* sAc = sA3[cur];
    AS3 const uint16_t* sWc = sW3[cur];
    #pragma unroll
    for (int kk = 0; kk < KC; kk += 32) {
      v16bf bfrag[2];
      #pragma unroll
      for (int tn = 0; tn < 2; ++tn) {
        int nrow = waveN * 32 + tn * 16 + lane15;
        bfrag[tn] = *(AS3 const v16bf*)(&sWc[nrow * LDT + kk + laneHalf * 16]);
      }
      #pragma unroll
      for (int tm = 0; tm < 4; ++tm) {
        int mrow = waveM * 64 + tm * 16 + lane15;
        AS3 const uint16_t* ap = &sAc[mrow * LDT + kk + laneHalf * 16];
        v8bf alo = *(AS3 const v8bf*)(ap);        // K = base .. base+7
        v8bf ahi = *(AS3 const v8bf*)(ap + 8);    // K = base+8 .. base+15
        v16bf afrag = __builtin_shufflevector(
            alo, ahi, 0,1,2,3,4,5,6,7,8,9,10,11,12,13,14,15);
        #pragma unroll
        for (int tn = 0; tn < 2; ++tn) {
          acc[tm][tn] = __builtin_amdgcn_wmma_f32_16x16x32_bf16(
              false, afrag, false, bfrag[tn], (short)0, acc[tm][tn],
              false, false);
        }
      }
    }

    // 4) dequant next W-tile into the other buffer (VALU co-exec w/ WMMA tail)
    if (haveNext) {
      const uint32_t* q = reinterpret_cast<const uint32_t*>(wq);
      #pragma unroll
      for (int i = 0; i < 16; ++i) wDst[nxt][i] = dq2(q[i]);
    }

    // 5) my async copies landed, then workgroup-wide visibility
    wait_async0();
    __syncthreads();
  }

  // ---- epilogue: fp4 bias dequant + bf16 store ----
  #pragma unroll
  for (int tn = 0; tn < 2; ++tn) {
    int n = blockN0 + waveN * 32 + tn * 16 + lane15;
    uint32_t pb  = (uint32_t)bp[n >> 1] & 0xFFu;
    uint32_t nib = (n & 1) ? (pb >> 4) : (pb & 0xFu);
    float bias = bf16bits_to_f32(nib2bf_mask(nib));
    #pragma unroll
    for (int tm = 0; tm < 4; ++tm) {
      int mbase = blockM0 + waveM * 64 + tm * 16 + laneHalf * 8;
      #pragma unroll
      for (int i = 0; i < 8; ++i) {
        float v = acc[tm][tn][i] + bias;   // C/D layout: VGPR i = row mbase+i
        out[(size_t)(mbase + i) * N_TOT + n] = f32_to_bf16_rne(v);
      }
    }
  }
}

extern "C" void kernel_launch(void* const* d_in, const int* in_sizes, int n_in,
                              void* d_out, int out_size, void* d_ws, size_t ws_size,
                              hipStream_t stream) {
  (void)in_sizes; (void)n_in; (void)out_size; (void)d_ws; (void)ws_size;
  const uint16_t* x  = (const uint16_t*)d_in[0];  // bf16 bits
  const int*      wp = (const int*)d_in[1];
  const int*      bp = (const int*)d_in[2];
  uint16_t*       out = (uint16_t*)d_out;         // bf16 bits

  dim3 grid(N_TOT / BN, M_TOT / BM);   // 86 x 4
  fp4_linear_wmma<<<grid, dim3(256), 0, stream>>>(x, wp, bp, out);
}
